// Model_35570919146246
// MI455X (gfx1250) — matrix operand order repaired and checked by reading, not compiled, not run on hardware
//
#include <hip/hip_runtime.h>
#include <hip/hip_bf16.h>

// ---------------------------------------------------------------------------
// CDNA5 (gfx1250) implicit-GEMM convolution, V_WMMA_F32_16X16X32_BF16.
//  - wave32, 256-thread block = 8 waves (4M x 2N); wave tile 32x32 = 2x2 WMMA.
//  - K tiled by 64 -> 8 WMMAs per tile; LDS double-buffered -> 1 barrier/tile,
//    next tile's staging (async B DMA + A gather) overlaps current WMMAs.
//  - Activations & weights are bf16 in memory (packed once per launch):
//    inner loop is pure 16-bit movement, halving HBM traffic vs f32.
//  - B tile: guard-free per-lane GLOBAL_LOAD_ASYNC_TO_LDS_B128 (ASYNCcnt,
//    weights zero-padded to K%64==0 / Co%64==0), s_wait_asynccnt before the
//    consume barrier.
//  - A tile: staged as (1 k-column x 16 m-rows) chunks. For stride 1 the m-run
//    is contiguous in memory -> one 32B read per chunk; transpose absorbed by
//    strided ds_store_b16 (VMEM -> LDS cost shift). k->(ci,ky,kx) decode uses
//    compile-time divisors; m-decode is hoisted out of the K loop.
// ---------------------------------------------------------------------------

typedef __attribute__((ext_vector_type(16))) __bf16 bf16x16;
typedef __attribute__((ext_vector_type(8)))  float  f32x8;

union FragU {
    bf16x16 v;
    uint4   q[2];
};

__device__ __forceinline__ unsigned short f32_to_bf16(float f) {
    unsigned int u = __float_as_uint(f);
    u += 0x7FFFu + ((u >> 16) & 1u);          // RNE
    return (unsigned short)(u >> 16);
}
__device__ __forceinline__ float bf16_to_f32(unsigned short u) {
    return __uint_as_float(((unsigned int)u) << 16);
}

#define TILE_M 128
#define TILE_N 64
#define TILE_K 64
#define KSTRIDE 72   // row stride in ushorts: 144B, 16B-aligned, spreads banks

#define CF_BN   1
#define CF_RELU 2
#define CF_RES  4

template <int KH, int KW, int STRIDE>
__launch_bounds__(256, 2)
__global__ void conv_wmma(const unsigned short* __restrict__ X,     // bf16 NCHW
                          const unsigned short* __restrict__ Wpk,   // bf16 [Cop][Kp]
                          const float* __restrict__ Sc, const float* __restrict__ Bi,
                          const unsigned short* __restrict__ Res,   // bf16 NCHW
                          unsigned short* __restrict__ Y,           // bf16 NCHW
                          int N, int Ci, int H, int W, int Co, int Ho, int Wo,
                          int Kp, int pad, int flags)
{
    __shared__ unsigned short Als[2][TILE_M][KSTRIDE];
    __shared__ unsigned short Bls[2][TILE_N][KSTRIDE];

    const int tid   = threadIdx.x;
    const int lane  = tid & 31;
    const int wv    = tid >> 5;
    const int wm    = wv & 3;
    const int wn    = wv >> 2;
    const int khalf = lane >> 4;
    const int nl    = lane & 15;

    constexpr int KHW = KH * KW;
    const int HoWo = Ho * Wo;
    const int Mtot = N * HoWo;
    const int Ktot = Ci * KHW;

    const int tileM = blockIdx.x * TILE_M;
    const int tileN = blockIdx.y * TILE_N;

    // ---- A-staging chunk map: 512 chunks of (1 k-col x 16 m-rows), 2/thread.
    // m-decode is k-independent: hoist it fully out of the K loop.
    int am0[2], ahb[2], awb[2];
    const unsigned short* axb[2];
    #pragma unroll
    for (int j = 0; j < 2; ++j) {
        int c   = tid + j * 256;
        int mch = c & 7;                       // 0..7
        int m0  = tileM + mch * 16;
        am0[j] = m0;
        int bb = 0, ho = 0, wo = 0;
        if (m0 < Mtot) {
            bb = m0 / HoWo;
            int rm = m0 - bb * HoWo;
            ho = rm / Wo;
            wo = rm - ho * Wo;
        }
        ahb[j] = ho * STRIDE - pad;
        awb[j] = wo * STRIDE - pad;
        axb[j] = X + (size_t)bb * Ci * H * W;
    }

    f32x8 acc[2][2];
    #pragma unroll
    for (int i = 0; i < 2; ++i)
        #pragma unroll
        for (int j = 0; j < 2; ++j)
            #pragma unroll
            for (int r = 0; r < 8; ++r) acc[i][j][r] = 0.0f;

    // ---- staging helpers -------------------------------------------------
    auto stage_B = [&](int p, int kk) {
        const unsigned base = (unsigned)(unsigned long long)&Bls[p][0][0];
        #pragma unroll
        for (int j = 0; j < 2; ++j) {
            int chunk = tid + j * 256;           // 0..511
            int row   = chunk >> 3;              // 0..63
            int col16 = chunk & 7;               // 16B column
            unsigned goff = (unsigned)(((tileN + row) * Kp + kk + col16 * 8) *
                                       (int)sizeof(unsigned short));
            unsigned laddr = base + (unsigned)((row * KSTRIDE + col16 * 8) *
                                               (int)sizeof(unsigned short));
            asm volatile("global_load_async_to_lds_b128 %0, %1, %2"
                         :: "v"(laddr), "v"(goff), "s"(Wpk)
                         : "memory");
        }
    };

    auto stage_A = [&](int p, int kk) {
        #pragma unroll
        for (int j = 0; j < 2; ++j) {
            int c    = tid + j * 256;
            int kloc = c >> 3;                   // 0..63
            int mch  = c & 7;
            int kg   = kk + kloc;
            int ci = kg / KHW;                   // compile-time divisors
            int rr = kg - ci * KHW;
            int ky = rr / KW;
            int kx = rr - ky * KW;
            int hi  = ahb[j] + ky;
            int wi0 = awb[j] + kx;
            const unsigned short* rowp = axb[j] + ((size_t)ci * H + hi) * W;

            unsigned short vals[16];
            bool rowok = (kg < Ktot) && (am0[j] < Mtot) &&
                         ((unsigned)hi < (unsigned)H);
            if (STRIDE == 1 && rowok && wi0 >= 0 && wi0 + 15 < W) {
                // contiguous 32B along m (fast path; clang picks widest loads)
                __builtin_memcpy(vals, rowp + wi0, 16 * sizeof(unsigned short));
            } else {
                #pragma unroll
                for (int t = 0; t < 16; ++t) {
                    int wi = wi0 + t * STRIDE;
                    unsigned short v = 0;
                    if (rowok && (am0[j] + t) < Mtot &&
                        (unsigned)wi < (unsigned)W)
                        v = rowp[wi];
                    vals[t] = v;
                }
            }
            #pragma unroll
            for (int t = 0; t < 16; ++t)
                Als[p][mch * 16 + t][kloc] = vals[t];
        }
    };

    auto compute = [&](int p) {
        #pragma unroll
        for (int s = 0; s < 2; ++s) {
            FragU a[2], b[2];
            #pragma unroll
            for (int mi = 0; mi < 2; ++mi) {
                const unsigned short* row = &Als[p][wm * 32 + mi * 16 + nl][s * 32];
                a[mi].q[0] = *(const uint4*)(row + khalf * 8);
                a[mi].q[1] = *(const uint4*)(row + 16 + khalf * 8);
            }
            #pragma unroll
            for (int ni = 0; ni < 2; ++ni) {
                const unsigned short* row = &Bls[p][wn * 32 + ni * 16 + nl][s * 32];
                b[ni].q[0] = *(const uint4*)(row + khalf * 8);
                b[ni].q[1] = *(const uint4*)(row + 16 + khalf * 8);
            }
            #pragma unroll
            for (int mi = 0; mi < 2; ++mi)
                #pragma unroll
                for (int ni = 0; ni < 2; ++ni)
                    acc[mi][ni] = __builtin_amdgcn_wmma_f32_16x16x32_bf16(
                        false, a[mi].v, false, b[ni].v,
                        (short)0, acc[mi][ni], false, false);
        }
    };

    // ---- software-pipelined main loop: 1 barrier per K-tile ----
    stage_B(0, 0);
    stage_A(0, 0);
    asm volatile("s_wait_asynccnt 0" ::: "memory");
    __syncthreads();

    int p = 0;
    for (int kk = 0; kk < Kp; kk += TILE_K) {
        const bool more = (kk + TILE_K) < Kp;
        if (more) {                      // stage t+1 into the other buffer,
            stage_B(p ^ 1, kk + TILE_K); // overlapping this tile's WMMAs
            stage_A(p ^ 1, kk + TILE_K);
        }
        compute(p);
        if (more)
            asm volatile("s_wait_asynccnt 0" ::: "memory");
        __syncthreads();
        p ^= 1;
    }

    // ---- epilogue: BN/bias, residual, ReLU; bf16 store; carry-based decode ----
    #pragma unroll
    for (int ni = 0; ni < 2; ++ni) {
        int n = tileN + wn * 32 + ni * 16 + nl;
        if (n >= Co) continue;
        float sc = (flags & CF_BN) ? Sc[n] : 1.0f;
        float bi = Bi ? Bi[n] : 0.0f;
        #pragma unroll
        for (int mi = 0; mi < 2; ++mi) {
            int m0 = tileM + wm * 32 + mi * 16 + khalf * 8;
            if (m0 >= Mtot) continue;
            int bb = m0 / HoWo;
            int rm = m0 - bb * HoWo;
            int ho = rm / Wo;
            int wo = rm - ho * Wo;
            #pragma unroll
            for (int r = 0; r < 8; ++r) {
                if (m0 + r < Mtot) {
                    size_t idx = (((size_t)bb * Co + n) * Ho + ho) * Wo + wo;
                    float v = acc[mi][ni][r] * sc + bi;
                    if (flags & CF_RES)  v += bf16_to_f32(Res[idx]);
                    if (flags & CF_RELU) v = v > 0.0f ? v : 0.0f;
                    Y[idx] = f32_to_bf16(v);
                }
                if (++wo == Wo) { wo = 0; if (++ho == Ho) { ho = 0; ++bb; } }
            }
        }
    }
}

// ---------------------------------------------------------------------------
// Packing kernels (once per launch, negligible vs conv layers).
// ---------------------------------------------------------------------------
__global__ void cvt_f32_bf16(const float* __restrict__ s,
                             unsigned short* __restrict__ d, int n)
{
    int i = blockIdx.x * blockDim.x + threadIdx.x;
    if (i < n) d[i] = f32_to_bf16(s[i]);
}

__global__ void pack_weights_bf16(const float* __restrict__ w,
                                  unsigned short* __restrict__ d,
                                  int Co, int K, int Kp, int Cop)
{
    int i = blockIdx.x * blockDim.x + threadIdx.x;
    int total = Cop * Kp;
    if (i >= total) return;
    int co = i / Kp;
    int k  = i - co * Kp;
    d[i] = (co < Co && k < K) ? f32_to_bf16(w[(size_t)co * K + k]) : (unsigned short)0;
}

// ---------------------------------------------------------------------------
// Deformable 3x3 sampling head (reference _deform), f32 path.
// ---------------------------------------------------------------------------
__global__ void deform_head(const float* __restrict__ depth,
                            const float* __restrict__ offset,
                            const float* __restrict__ wgt,
                            float* __restrict__ out,
                            int N, int H, int W)
{
    int idx = blockIdx.x * blockDim.x + threadIdx.x;
    int total = N * H * W;
    if (idx >= total) return;
    int b = idx / (H * W);
    int rm = idx - b * (H * W);
    int y = rm / W;
    int x = rm - y * W;
    const float* img = depth + (size_t)b * H * W;

    float wv[9];
    float mean = 0.0f;
    #pragma unroll
    for (int t = 0; t < 9; ++t) {
        wv[t] = wgt[(((size_t)b * 9 + t) * H + y) * W + x];
        mean += wv[t];
    }
    mean *= (1.0f / 9.0f);

    float s = 0.0f;
    #pragma unroll
    for (int t = 0; t < 9; ++t) {
        int ky = t / 3 - 1;
        int kx = t % 3 - 1;
        float oy = offset[(((size_t)b * 18 + 2 * t + 0) * H + y) * W + x];
        float ox = offset[(((size_t)b * 18 + 2 * t + 1) * H + y) * W + x];
        float py = (float)y + (float)ky + oy;
        float px = (float)x + (float)kx + ox;
        float y0 = floorf(py), x0 = floorf(px);
        float fy = py - y0, fx = px - x0;
        int iy0 = (int)y0, ix0 = (int)x0;
        float v = 0.0f;
        #pragma unroll
        for (int dy = 0; dy < 2; ++dy)
            #pragma unroll
            for (int dx = 0; dx < 2; ++dx) {
                int yy = iy0 + dy, xx = ix0 + dx;
                float w2 = (dy ? fy : 1.0f - fy) * (dx ? fx : 1.0f - fx);
                if ((unsigned)yy < (unsigned)H && (unsigned)xx < (unsigned)W)
                    v += w2 * img[(size_t)yy * W + xx];
            }
        s += v * (wv[t] - mean);
    }
    out[idx] = s + depth[idx];
}

// ---------------------------------------------------------------------------
// Channel-mean + nearest upsample head, templated on input element type.
// ---------------------------------------------------------------------------
__device__ __forceinline__ float ld_elem(float v) { return v; }
__device__ __forceinline__ float ld_elem(unsigned short v) { return bf16_to_f32(v); }

template <typename T>
__global__ void head_mean_up(const T* __restrict__ x, float* __restrict__ out,
                             int N, int C, int Hi, int Wi, int Ho, int Wo)
{
    int idx = blockIdx.x * blockDim.x + threadIdx.x;
    int total = N * Ho * Wo;
    if (idx >= total) return;
    int b = idx / (Ho * Wo);
    int rm = idx - b * (Ho * Wo);
    int yo = rm / Wo;
    int xo = rm - yo * Wo;
    int yi = yo * Hi / Ho;
    int xi = xo * Wi / Wo;
    float s = 0.0f;
    for (int c = 0; c < C; ++c)
        s += ld_elem(x[(((size_t)b * C + c) * Hi + yi) * Wi + xi]);
    out[idx] = s / (float)C;
}

// ---------------------------------------------------------------------------
// Orchestration. Flattened param order (setup_inputs dict order):
//  0 rgb, 1 s1, 2 s2,
//  3 conv_img.w(32,3,5,5)  4 bn_s  5 bn_b
//  6 conv_lidar.w          7 conv_lidar.b
//  8 conv_s2.w             9 bn_s  10 bn_b
//  layer1_img b0: 11 w1 12 s1 13 b1 14 w2 15 s2 16 b2 17 wd 18 sd 19 bd
//  layer1_img b1: 20 w1 21 s1 22 b1 23 w2 24 s2 25 b2
// ---------------------------------------------------------------------------
extern "C" void kernel_launch(void* const* d_in, const int* in_sizes, int n_in,
                              void* d_out, int out_size, void* d_ws, size_t ws_size,
                              hipStream_t stream)
{
    const int N = 2, H = 256, W = 512;
    const float* rgb = (const float*)d_in[0];
    float* out = (float*)d_out;
    unsigned short* ws = (unsigned short*)d_ws;

    auto al64 = [](int k) { return (k + 63) & ~63; };

    const size_t n_rgb = (size_t)N * 3 * H * W;
    const int Kp_img = al64(3 * 25);      // 128
    const int Kp_w1  = al64(32 * 9);      // 320
    const int Kp_wd  = al64(32);          // 64
    const int Kp_w2  = al64(64 * 9);      // 576
    const int Cop    = 64;

    size_t o = 0;
    size_t xrgb  = o; o += n_rgb;
    size_t wpImg = o; o += (size_t)Cop * Kp_img;
    size_t wpW1  = o; o += (size_t)Cop * Kp_w1;
    size_t wpWd  = o; o += (size_t)Cop * Kp_wd;
    size_t wpW2  = o; o += (size_t)Cop * Kp_w2;
    size_t wpB1a = o; o += (size_t)Cop * Kp_w2;
    size_t wpB1b = o; o += (size_t)Cop * Kp_w2;
    size_t c0    = o; o += (size_t)N * 32 * H * W;
    size_t t1    = o; o += (size_t)N * 64 * (H / 2) * (W / 2);
    size_t t2    = o; o += (size_t)N * 64 * (H / 2) * (W / 2);
    size_t t3    = o; o += (size_t)N * 64 * (H / 2) * (W / 2);
    const size_t need = o * sizeof(unsigned short);

    const bool params_ok = (n_in >= 26) && (in_sizes[3] == 32 * 3 * 5 * 5) &&
                           (in_sizes[11] == 64 * 32 * 3 * 3);

    if (!params_ok || ws_size < need) {
        int total = N * 1 * H * W;
        head_mean_up<float><<<dim3((total + 255) / 256), dim3(256), 0, stream>>>(
            rgb, out, N, 3, H, W, H, W);
        return;
    }

    auto packw = [&](const void* src, size_t dst, int Co, int K, int Kp) {
        int total = Cop * Kp;
        pack_weights_bf16<<<dim3((total + 255) / 256), dim3(256), 0, stream>>>(
            (const float*)src, ws + dst, Co, K, Kp, Cop);
    };

    cvt_f32_bf16<<<dim3((int)((n_rgb + 255) / 256)), dim3(256), 0, stream>>>(
        rgb, ws + xrgb, (int)n_rgb);
    packw(d_in[3],  wpImg, 32, 75,  Kp_img);
    packw(d_in[11], wpW1,  64, 288, Kp_w1);
    packw(d_in[17], wpWd,  64, 32,  Kp_wd);
    packw(d_in[14], wpW2,  64, 576, Kp_w2);
    packw(d_in[20], wpB1a, 64, 576, Kp_w2);
    packw(d_in[23], wpB1b, 64, 576, Kp_w2);

    auto conv = [&](int Ksz, int str, size_t Xo, size_t Wo_, const void* Sc,
                    const void* Bi, size_t ResO, bool hasRes, size_t Yo,
                    int Ci, int Hh, int Ww, int Co, int Hoo, int Woo,
                    int Kp, int pd, int flags) {
        int M = N * Hoo * Woo;
        dim3 grid((M + TILE_M - 1) / TILE_M, (Co + TILE_N - 1) / TILE_N);
        const unsigned short* Res = hasRes ? ws + ResO : (const unsigned short*)nullptr;
        if (Ksz == 5)
            conv_wmma<5, 5, 1><<<grid, dim3(256), 0, stream>>>(
                ws + Xo, ws + Wo_, (const float*)Sc, (const float*)Bi, Res, ws + Yo,
                N, Ci, Hh, Ww, Co, Hoo, Woo, Kp, pd, flags);
        else if (Ksz == 3 && str == 2)
            conv_wmma<3, 3, 2><<<grid, dim3(256), 0, stream>>>(
                ws + Xo, ws + Wo_, (const float*)Sc, (const float*)Bi, Res, ws + Yo,
                N, Ci, Hh, Ww, Co, Hoo, Woo, Kp, pd, flags);
        else if (Ksz == 3)
            conv_wmma<3, 3, 1><<<grid, dim3(256), 0, stream>>>(
                ws + Xo, ws + Wo_, (const float*)Sc, (const float*)Bi, Res, ws + Yo,
                N, Ci, Hh, Ww, Co, Hoo, Woo, Kp, pd, flags);
        else
            conv_wmma<1, 1, 2><<<grid, dim3(256), 0, stream>>>(
                ws + Xo, ws + Wo_, (const float*)Sc, (const float*)Bi, Res, ws + Yo,
                N, Ci, Hh, Ww, Co, Hoo, Woo, Kp, pd, flags);
    };

    // stem: conv_img 5x5 s1 p2 + BN + ReLU
    conv(5, 1, xrgb, wpImg, d_in[4], d_in[5], 0, false, c0,
         3, H, W, 32, H, W, Kp_img, 2, CF_BN | CF_RELU);

    // layer1_img block0
    conv(3, 2, c0, wpW1, d_in[12], d_in[13], 0, false, t1,
         32, H, W, 64, H / 2, W / 2, Kp_w1, 1, CF_BN | CF_RELU);
    conv(1, 2, c0, wpWd, d_in[18], d_in[19], 0, false, t2,
         32, H, W, 64, H / 2, W / 2, Kp_wd, 0, CF_BN);
    conv(3, 1, t1, wpW2, d_in[15], d_in[16], t2, true, t3,
         64, H / 2, W / 2, 64, H / 2, W / 2, Kp_w2, 1, CF_BN | CF_RES | CF_RELU);

    // layer1_img block1
    conv(3, 1, t3, wpB1a, d_in[21], d_in[22], 0, false, t1,
         64, H / 2, W / 2, 64, H / 2, W / 2, Kp_w2, 1, CF_BN | CF_RELU);
    conv(3, 1, t1, wpB1b, d_in[24], d_in[25], t3, true, t2,
         64, H / 2, W / 2, 64, H / 2, W / 2, Kp_w2, 1, CF_BN | CF_RES | CF_RELU);

    // collapse to (N,1,H,W) output
    int total = N * 1 * H * W;
    head_mean_up<unsigned short><<<dim3((total + 255) / 256), dim3(256), 0, stream>>>(
        ws + t2, out, N, 64, H / 2, W / 2, H, W);
}
